// Ins_29317446762502
// MI455X (gfx1250) — matrix operand order repaired and checked by reading, not compiled, not run on hardware
//
#include <hip/hip_runtime.h>
#include <hip/hip_bf16.h>

typedef __attribute__((ext_vector_type(16))) _Float16 v16h;
typedef __attribute__((ext_vector_type(8)))  float    v8f;

#define NINS 8
#define TK   8
#define DDIM 512
#define NCLS 2
#define SCAN_BLOCKS  64
#define SCAN_THREADS 256
#define NEG_INF (-__builtin_inff())

// ---------------------------------------------------------------------------
// Kernel 1: streaming single-pass scan of A[:, 0, bag_label].
// Each thread keeps branchless register-resident top-8 (side 0) and
// bottom-8 (side 1, stored negated) candidate lists; block reduces its
// 256*8 candidates per side to 8 winners via 8 rounds of parallel argmax.
// ---------------------------------------------------------------------------
__global__ void __launch_bounds__(SCAN_THREADS)
topk_scan_kernel(const float* __restrict__ A, const int* __restrict__ bag_label,
                 int N, float* __restrict__ blkmax_v, int* __restrict__ blkmax_i,
                 float* __restrict__ blkmin_v, int* __restrict__ blkmin_i)
{
    __shared__ float cv[2][SCAN_THREADS * TK];
    __shared__ int   ci[2][SCAN_THREADS * TK];
    __shared__ float rv[SCAN_THREADS];
    __shared__ int   ri[SCAN_THREADS];
    __shared__ int   rs[SCAN_THREADS];

    const int tid    = threadIdx.x;
    const int gid    = blockIdx.x * SCAN_THREADS + tid;
    const int stride = gridDim.x * SCAN_THREADS;
    const int bl     = bag_label[0];

    float tv[2][TK];
    int   ti[2][TK];
#pragma unroll
    for (int s = 0; s < 2; ++s)
#pragma unroll
        for (int j = 0; j < TK; ++j) { tv[s][j] = NEG_INF; ti[s][j] = 0x7FFFFFFF; }

    for (int i = gid; i < N; i += stride) {
        const float v = A[i * NCLS + bl];
        float vals[2] = { v, -v };
#pragma unroll
        for (int s = 0; s < 2; ++s) {
            const float x = vals[s];
            // branchless sorted-descending insertion (j descending reads old vals)
#pragma unroll
            for (int j = TK - 1; j >= 1; --j) {
                const bool gt  = x > tv[s][j];
                const bool gtp = x > tv[s][j - 1];
                const float nv = gt ? (gtp ? tv[s][j - 1] : x) : tv[s][j];
                const int   ni = gt ? (gtp ? ti[s][j - 1] : i) : ti[s][j];
                tv[s][j] = nv; ti[s][j] = ni;
            }
            if (x > tv[s][0]) { tv[s][0] = x; ti[s][0] = i; }
        }
    }

#pragma unroll
    for (int s = 0; s < 2; ++s)
#pragma unroll
        for (int j = 0; j < TK; ++j) {
            cv[s][tid * TK + j] = tv[s][j];
            ci[s][tid * TK + j] = ti[s][j];
        }
    __syncthreads();

    for (int s = 0; s < 2; ++s) {
        for (int r = 0; r < TK; ++r) {
            float bv = NEG_INF; int bi = 0x7FFFFFFF; int bs = -1;
            for (int j = tid; j < SCAN_THREADS * TK; j += SCAN_THREADS) {
                const float v = cv[s][j]; const int ii = ci[s][j];
                if (v > bv || (v == bv && ii < bi)) { bv = v; bi = ii; bs = j; }
            }
            rv[tid] = bv; ri[tid] = bi; rs[tid] = bs;
            __syncthreads();
            for (int off = SCAN_THREADS / 2; off > 0; off >>= 1) {
                if (tid < off) {
                    const float v2 = rv[tid + off]; const int i2 = ri[tid + off];
                    if (v2 > rv[tid] || (v2 == rv[tid] && i2 < ri[tid])) {
                        rv[tid] = v2; ri[tid] = i2; rs[tid] = rs[tid + off];
                    }
                }
                __syncthreads();
            }
            if (tid == 0) {
                float* ov = (s == 0) ? blkmax_v : blkmin_v;
                int*   oi = (s == 0) ? blkmax_i : blkmin_i;
                ov[blockIdx.x * TK + r] = rv[0];
                oi[blockIdx.x * TK + r] = ri[0];
                cv[s][rs[0]] = NEG_INF;
                ci[s][rs[0]] = 0x7FFFFFFF;
            }
            __syncthreads();
        }
    }
}

// ---------------------------------------------------------------------------
// Kernel 2: merge 64 blocks * 8 candidates per side -> final 16 indices.
// Also writes the constant instance labels (as floats) to d_out[0..15].
// ---------------------------------------------------------------------------
__global__ void __launch_bounds__(SCAN_THREADS)
topk_final_kernel(const float* __restrict__ blkmax_v, const int* __restrict__ blkmax_i,
                  const float* __restrict__ blkmin_v, const int* __restrict__ blkmin_i,
                  int nblk, int* __restrict__ idx16, float* __restrict__ out)
{
    __shared__ float cv[2][SCAN_BLOCKS * TK];
    __shared__ int   ci[2][SCAN_BLOCKS * TK];
    __shared__ float rv[SCAN_THREADS];
    __shared__ int   ri[SCAN_THREADS];
    __shared__ int   rs[SCAN_THREADS];

    const int tid = threadIdx.x;
    const int tot = nblk * TK;
    for (int j = tid; j < tot; j += SCAN_THREADS) {
        cv[0][j] = blkmax_v[j]; ci[0][j] = blkmax_i[j];
        cv[1][j] = blkmin_v[j]; ci[1][j] = blkmin_i[j];
    }
    __syncthreads();

    for (int s = 0; s < 2; ++s) {
        for (int r = 0; r < TK; ++r) {
            float bv = NEG_INF; int bi = 0x7FFFFFFF; int bs = -1;
            for (int j = tid; j < tot; j += SCAN_THREADS) {
                const float v = cv[s][j]; const int ii = ci[s][j];
                if (v > bv || (v == bv && ii < bi)) { bv = v; bi = ii; bs = j; }
            }
            rv[tid] = bv; ri[tid] = bi; rs[tid] = bs;
            __syncthreads();
            for (int off = SCAN_THREADS / 2; off > 0; off >>= 1) {
                if (tid < off) {
                    const float v2 = rv[tid + off]; const int i2 = ri[tid + off];
                    if (v2 > rv[tid] || (v2 == rv[tid] && i2 < ri[tid])) {
                        rv[tid] = v2; ri[tid] = i2; rs[tid] = rs[tid + off];
                    }
                }
                __syncthreads();
            }
            if (tid == 0) {
                idx16[s * NINS + r] = ri[0];
                cv[s][rs[0]] = NEG_INF;
                ci[s][rs[0]] = 0x7FFFFFFF;
            }
            __syncthreads();
        }
    }
    if (tid < 16) out[tid] = (tid < NINS) ? 1.0f : 0.0f;   // ins_labels
}

// ---------------------------------------------------------------------------
// Kernel 3: single-wave WMMA head.  D(16x16,f32) += A(16x32,f16) x B(32x16,f16)
// over K=512 (16 wmma issues, unrolled x4).
// W is staged once into LDS as a zero-padded 16x512 f16 matrix so the B
// fragment is one contiguous 32B LDS read (no exec-masked loads); the A
// fragment is four aligned float4 global loads per step.
// Fragment layouts used here (16-bit, same K order for A and B):
//   A: lane L holds M=L&15; half j -> K = j+16g, g=L>>4
//   B: lane L holds N=L&15; half j -> K = j+16g
//   C/D: lane L, elem j -> M = j+8g, N = L&15
// ---------------------------------------------------------------------------
__global__ void __launch_bounds__(32)
wmma_head_kernel(const float* __restrict__ h, const float* __restrict__ W,
                 const float* __restrict__ bias, const int* __restrict__ idx16,
                 float* __restrict__ out)
{
    __shared__ _Float16 Wt[16][DDIM];    // [n][k], rows NCLS..15 are zero
    __shared__ float lds_lu[16][2];

    const int lane = threadIdx.x;
    const int g    = lane >> 4;
    const int n    = lane & 15;

    // ---- stage W into LDS (uniform loops, all 32 lanes active) ----
    {
        // zero fill 16*512 halves = 4096 dwords, 128 dwords per lane
        unsigned int* wz = (unsigned int*)&Wt[0][0];
#pragma unroll
        for (int j = 0; j < 128; ++j) wz[lane + j * 32] = 0u;
        __syncthreads();
        // rows 0..NCLS-1: Wt[c][k] = (f16) W[k*NCLS + c]
        for (int idx = lane; idx < DDIM * NCLS; idx += 32) {
            const int k = idx >> 1;
            const int c = idx & 1;
            Wt[c][k] = (_Float16)W[idx];
        }
        __syncthreads();
    }

    const int row = idx16[n];
    const float* __restrict__ hrow = h + (size_t)row * DDIM;

    v8f acc = {};
#pragma unroll 4
    for (int k0 = 0; k0 < DDIM; k0 += 32) {
        // A fragment: 4 aligned float4 loads (one contiguous K run of 16, K = k0+16g..+15)
        const float4 alo0 = *(const float4*)(hrow + k0 + 16 * g);
        const float4 alo1 = *(const float4*)(hrow + k0 + 16 * g + 4);
        const float4 ahi0 = *(const float4*)(hrow + k0 + 16 * g + 8);
        const float4 ahi1 = *(const float4*)(hrow + k0 + 16 * g + 12);
        v16h a;
        a[0]  = (_Float16)alo0.x; a[1]  = (_Float16)alo0.y;
        a[2]  = (_Float16)alo0.z; a[3]  = (_Float16)alo0.w;
        a[4]  = (_Float16)alo1.x; a[5]  = (_Float16)alo1.y;
        a[6]  = (_Float16)alo1.z; a[7]  = (_Float16)alo1.w;
        a[8]  = (_Float16)ahi0.x; a[9]  = (_Float16)ahi0.y;
        a[10] = (_Float16)ahi0.z; a[11] = (_Float16)ahi0.w;
        a[12] = (_Float16)ahi1.x; a[13] = (_Float16)ahi1.y;
        a[14] = (_Float16)ahi1.z; a[15] = (_Float16)ahi1.w;

        // B fragment: one contiguous 32B LDS read (16 halves, K = k0+16g..+15)
        const v16h b = *(const v16h*)(&Wt[n][k0 + 16 * g]);

        acc = __builtin_amdgcn_wmma_f32_16x16x32_f16(
            /*neg_a=*/false, a, /*neg_b=*/false, b,
            /*c_mod=*/(short)0, acc, /*reuse_a=*/false, /*reuse_b=*/false);
    }

    if (n < NCLS) {
        const float bn = bias[n];
#pragma unroll
        for (int j = 0; j < 8; ++j) {
            const int m = j + 8 * g;
            const float lu = acc[j] + bn;
            out[16 + m * NCLS + n] = lu;   // logits_unnorm
            lds_lu[m][n] = lu;
        }
    }
    __syncthreads();

    if (lane < 16) {
        const float l0 = lds_lu[lane][0];
        const float l1 = lds_lu[lane][1];
        const float mx = fmaxf(l0, l1);
        const float e0 = __expf(l0 - mx);
        const float e1 = __expf(l1 - mx);
        const float inv = 1.0f / (e0 + e1);
        out[48 + lane * NCLS + 0] = e0 * inv;   // softmax
        out[48 + lane * NCLS + 1] = e1 * inv;
    }
}

// ---------------------------------------------------------------------------
extern "C" void kernel_launch(void* const* d_in, const int* in_sizes, int n_in,
                              void* d_out, int out_size, void* d_ws, size_t ws_size,
                              hipStream_t stream) {
    const int*   bag = (const int*)d_in[0];
    const float* h   = (const float*)d_in[1];
    const float* A   = (const float*)d_in[2];
    const float* W   = (const float*)d_in[3];
    const float* b   = (const float*)d_in[4];
    float* out = (float*)d_out;
    const int N = in_sizes[2] / NCLS;   // A is (N,1,NCLS)

    char* ws = (char*)d_ws;
    float* blkmax_v = (float*)(ws + 0);
    int*   blkmax_i = (int*)  (ws + 2048);
    float* blkmin_v = (float*)(ws + 4096);
    int*   blkmin_i = (int*)  (ws + 6144);
    int*   idx16    = (int*)  (ws + 8192);

    topk_scan_kernel<<<SCAN_BLOCKS, SCAN_THREADS, 0, stream>>>(
        A, bag, N, blkmax_v, blkmax_i, blkmin_v, blkmin_i);
    topk_final_kernel<<<1, SCAN_THREADS, 0, stream>>>(
        blkmax_v, blkmax_i, blkmin_v, blkmin_i, SCAN_BLOCKS, idx16, out);
    wmma_head_kernel<<<1, 32, 0, stream>>>(h, W, b, idx16, out);
}
